// MoeLayer_38508676776577
// MI455X (gfx1250) — matrix operand order repaired and checked by reading, not compiled, not run on hardware
//
#include <hip/hip_runtime.h>
#include <hip/hip_bf16.h>

#define E_EXP 8
#define CAP 640
#define NTOK 4096
#define DMODEL 1024
#define FFN 4096

#define TM 128
#define TK 32
#define PITCH 40               // halfwords; rows are 80B -> 16B aligned, conflict-free frag reads

typedef __attribute__((ext_vector_type(16))) __bf16 v16bf;
typedef __attribute__((ext_vector_type(8)))  float  v8f;

__device__ __forceinline__ unsigned short f2bf(float f) {
    unsigned u = __float_as_uint(f);
    unsigned r = u + 0x7FFFu + ((u >> 16) & 1u);   // round-to-nearest-even
    return (unsigned short)(r >> 16);
}

__device__ __forceinline__ float gelu_tanh(float x) {
    float x3 = x * x * x;
    return 0.5f * x * (1.0f + tanhf(0.7978845608028654f * (x + 0.044715f * x3)));
}

// CDNA5 async global->LDS copy (ASYNCcnt tracked). lds_off = LDS byte offset
// (= low 32 bits of the flat shared-memory address, per ISA LDS aperture rule).
__device__ __forceinline__ void async_ld_b128(unsigned lds_off, const void* gptr) {
    asm volatile("global_load_async_to_lds_b128 %0, %1, off"
                 :: "v"(lds_off), "v"(gptr)
                 : "memory");
}
__device__ __forceinline__ void wait_async0() {
    asm volatile("s_wait_asynccnt 0x0" ::: "memory");
}

// ---------------- transpose + fp32 -> bf16 weight conversion ----------------
// src: fp32 [E][K][N] row-major  ->  dst: bf16 [E][N][K] row-major
__global__ __launch_bounds__(256) void cvt_t_bf16(const float* __restrict__ src,
                                                  unsigned short* __restrict__ dst,
                                                  int K, int N) {
    __shared__ float tile[64][65];
    const int e  = blockIdx.z;
    const int n0 = blockIdx.x * 64;
    const int k0 = blockIdx.y * 64;
    const float*    se = src + (size_t)e * K * N;
    unsigned short* de = dst + (size_t)e * N * K;
    const int t  = threadIdx.x;
    const int tr = t >> 4;            // 0..15
    const int tc = (t & 15) * 4;      // 0..60
#pragma unroll
    for (int r = 0; r < 4; ++r) {
        int row = tr + r * 16;        // k within tile
        float4 x = *(const float4*)(se + (size_t)(k0 + row) * N + n0 + tc);
        tile[row][tc + 0] = x.x; tile[row][tc + 1] = x.y;
        tile[row][tc + 2] = x.z; tile[row][tc + 3] = x.w;
    }
    __syncthreads();
#pragma unroll
    for (int r = 0; r < 4; ++r) {
        int nrow = tr + r * 16;       // n within tile
        unsigned lo = (unsigned)f2bf(tile[tc + 0][nrow]) |
                      ((unsigned)f2bf(tile[tc + 1][nrow]) << 16);
        unsigned hi = (unsigned)f2bf(tile[tc + 2][nrow]) |
                      ((unsigned)f2bf(tile[tc + 3][nrow]) << 16);
        *(uint2*)(de + (size_t)(n0 + nrow) * K + k0 + tc) = make_uint2(lo, hi);
    }
}

// ---------------- gate: logits -> softmax top-1 ----------------
__global__ __launch_bounds__(256) void gate_kernel(const float* __restrict__ tokens,
                                                   const float* __restrict__ gw,
                                                   const float* __restrict__ gb,
                                                   int* __restrict__ eidx,
                                                   float* __restrict__ gp) {
    int wave = threadIdx.x >> 5;
    int lane = threadIdx.x & 31;
    int n = blockIdx.x * 8 + wave;           // grid = NTOK/8, always in range
    const float* tp = tokens + (size_t)n * DMODEL;
    float acc[E_EXP];
#pragma unroll
    for (int e = 0; e < E_EXP; ++e) acc[e] = 0.0f;
    for (int d = lane; d < DMODEL; d += 32) {
        float x = tp[d];
        const float* wr = gw + (size_t)d * E_EXP;
#pragma unroll
        for (int e = 0; e < E_EXP; ++e) acc[e] += x * wr[e];
    }
#pragma unroll
    for (int e = 0; e < E_EXP; ++e) {
#pragma unroll
        for (int off = 16; off > 0; off >>= 1) acc[e] += __shfl_xor(acc[e], off, 32);
    }
    if (lane == 0) {
        float best = -1e30f; int bi = 0;
#pragma unroll
        for (int e = 0; e < E_EXP; ++e) {
            acc[e] += gb[e];
            if (acc[e] > best) { best = acc[e]; bi = e; }
        }
        float s = 0.0f;
#pragma unroll
        for (int e = 0; e < E_EXP; ++e) s += __expf(acc[e] - best);
        eidx[n] = bi;
        gp[n] = 1.0f / s;                    // softmax prob of the argmax
    }
}

// ---------------- router scan (exact cumsum order) ----------------
__global__ void init_slot(int* __restrict__ slot) {
    int i = blockIdx.x * 256 + threadIdx.x;
    if (i < E_EXP * CAP) slot[i] = -1;
}

__global__ void scan_kernel(const int* __restrict__ eidx,
                            int* __restrict__ pos,
                            int* __restrict__ slot) {
    int lane = threadIdx.x;                  // single wave32
    int cnt[E_EXP];
#pragma unroll
    for (int e = 0; e < E_EXP; ++e) cnt[e] = 0;
    unsigned below = (1u << lane) - 1u;
    for (int it = 0; it < NTOK / 32; ++it) {
        int n = it * 32 + lane;
        int e = eidx[n];
        int p = 0;
#pragma unroll
        for (int ex = 0; ex < E_EXP; ++ex) {
            unsigned m = (unsigned)__ballot(e == ex);
            if (e == ex) p = cnt[ex] + __popc(m & below);
            cnt[ex] += __popc(m);
        }
        pos[n] = p;
        if (p < CAP) slot[(size_t)e * CAP + p] = n;
    }
}

// ---------------- dispatch gather (fp32 tokens -> bf16 slots) ----------------
__global__ __launch_bounds__(256) void gather_kernel(const int* __restrict__ slot,
                                                     const float* __restrict__ tokens,
                                                     unsigned short* __restrict__ disp) {
    int s = blockIdx.x;                      // 0 .. E*CAP-1
    int t = threadIdx.x;                     // 256 threads x 4 elems = 1024
    int tok = slot[s];
    uint2 val = make_uint2(0u, 0u);
    if (tok >= 0) {
        float4 x = ((const float4*)(tokens + (size_t)tok * DMODEL))[t];
        val.x = (unsigned)f2bf(x.x) | ((unsigned)f2bf(x.y) << 16);
        val.y = (unsigned)f2bf(x.z) | ((unsigned)f2bf(x.w) << 16);
    }
    ((uint2*)(disp + (size_t)s * DMODEL))[t] = val;
}

// ---------------- tiled bf16 WMMA GEMM (per-expert batched) ----------------
// A:  [E][M][K] bf16 row-major (activations)
// BT: [E][N][K] bf16 row-major (pre-transposed weights)
// WN = 16-col fragments per wave (2 -> TN=128, 4 -> TN=256)
// GELU=true  -> C bf16 [E][M][N] = gelu(acc + bias)
// GELU=false -> C f32  [E][M][N] = acc + bias
template <int WN, bool GELU>
__global__ __launch_bounds__(256) void moe_gemm(const unsigned short* __restrict__ A,
                                                const unsigned short* __restrict__ BT,
                                                const float* __restrict__ bias,
                                                void* __restrict__ Cv,
                                                int M, int N, int K) {
    constexpr int TN   = 64 * WN;            // 4 waves across N
    constexpr int ASZ  = TM * PITCH;         // halfwords per A buffer
    constexpr int BSZ  = TN * PITCH;         // halfwords per B buffer
    __shared__ __align__(16) unsigned short As[2 * ASZ];
    __shared__ __align__(16) unsigned short Bs[2 * BSZ];   // [n][k]

    const int e  = blockIdx.z;
    const int n0 = blockIdx.x * TN;
    const int m0 = blockIdx.y * TM;
    const unsigned short* Ae  = A  + (size_t)e * M * K;
    const unsigned short* BTe = BT + (size_t)e * N * K;
    const float*          be  = bias + (size_t)e * N;

    const int tid  = threadIdx.x;
    const int lane = tid & 31;
    const int wave = tid >> 5;
    const int wm = wave & 1;                 // M half (64 rows)
    const int wn = wave >> 1;                // N quarter (WN*16 cols)
    const int lr = lane & 15;
    const int lh = lane >> 4;

    // cooperative tile-load indices: rows of 32 halfwords, 2 threads/row
    const int row  = tid >> 1;               // 0..127
    const int colb = (tid & 1) << 4;         // 0 / 16 (halfwords)
    const unsigned short* Ag = Ae  + (size_t)(m0 + row) * K + colb;
    const unsigned short* Bg = BTe + (size_t)(n0 + row) * K + colb;

    v8f acc[4][WN];
#pragma unroll
    for (int mi = 0; mi < 4; ++mi)
#pragma unroll
        for (int ni = 0; ni < WN; ++ni)
#pragma unroll
            for (int j = 0; j < 8; ++j) acc[mi][ni][j] = 0.0f;

    auto tile_async = [&](int k0, int buf) {
        const unsigned short* ga = Ag + k0;
        unsigned ao = (unsigned)(uintptr_t)&As[buf * ASZ + row * PITCH + colb];
        async_ld_b128(ao,      ga);
        async_ld_b128(ao + 16, ga + 8);
#pragma unroll
        for (int r = 0; r < TN / 128; ++r) {
            const unsigned short* gb = Bg + (size_t)r * 128 * K + k0;
            unsigned bo = (unsigned)(uintptr_t)
                &Bs[buf * BSZ + (r * 128 + row) * PITCH + colb];
            async_ld_b128(bo,      gb);
            async_ld_b128(bo + 16, gb + 8);
        }
    };

    const int nsteps = K / TK;
    tile_async(0, 0);
    wait_async0();
    __syncthreads();

    for (int s = 0; s < nsteps; ++s) {
        const int cur = s & 1;
        if (s + 1 < nsteps) tile_async((s + 1) * TK, (s + 1) & 1);  // overlap with math

        union { v16bf v; uint4 q[2]; } afr[4], bfr[WN];
#pragma unroll
        for (int mi = 0; mi < 4; ++mi) {
            // A frag 16x32: lane<16 -> K {0..7,16..23}; lane>=16 -> K {8..15,24..31}
            const unsigned short* ap =
                &As[cur * ASZ + (wm * 64 + mi * 16 + lr) * PITCH + lh * 8];
            afr[mi].q[0] = *(const uint4*)ap;
            afr[mi].q[1] = *(const uint4*)(ap + 16);
        }
#pragma unroll
        for (int ni = 0; ni < WN; ++ni) {
            // B frag 32x16: lane<16 -> K {0..7,16..23}; lane>=16 -> K {8..15,24..31}
            const unsigned short* bp =
                &Bs[cur * BSZ + (wn * WN * 16 + ni * 16 + lr) * PITCH + lh * 8];
            bfr[ni].q[0] = *(const uint4*)bp;
            bfr[ni].q[1] = *(const uint4*)(bp + 16);
        }
#pragma unroll
        for (int mi = 0; mi < 4; ++mi)
#pragma unroll
            for (int ni = 0; ni < WN; ++ni)
                acc[mi][ni] = __builtin_amdgcn_wmma_f32_16x16x32_bf16(
                    false, afr[mi].v, false, bfr[ni].v,
                    (short)0, acc[mi][ni], false, false);

        wait_async0();                        // next tile landed during the math
        __syncthreads();
    }

    // epilogue: C layout -> col = lr, row = lh*8 + j
    const int mb = m0 + wm * 64;
    const int nb = n0 + wn * WN * 16;
#pragma unroll
    for (int mi = 0; mi < 4; ++mi) {
#pragma unroll
        for (int ni = 0; ni < WN; ++ni) {
#pragma unroll
            for (int j = 0; j < 8; ++j) {
                int r = mb + mi * 16 + lh * 8 + j;
                int c = nb + ni * 16 + lr;
                float v = acc[mi][ni][j] + be[c];
                if (GELU) {
                    ((unsigned short*)Cv)[(size_t)e * M * N + (size_t)r * N + c] =
                        f2bf(gelu_tanh(v));
                } else {
                    ((float*)Cv)[(size_t)e * M * N + (size_t)r * N + c] = v;
                }
            }
        }
    }
}

// ---------------- combine ----------------
__global__ __launch_bounds__(256) void combine_kernel(const int* __restrict__ eidx,
                                                      const int* __restrict__ pos,
                                                      const float* __restrict__ gp,
                                                      const float* __restrict__ eout,
                                                      float* __restrict__ out) {
    int n = blockIdx.x;
    int t = threadIdx.x;
    int e = eidx[n];
    int p = pos[n];
    float g = gp[n];
    float4 v = make_float4(0.f, 0.f, 0.f, 0.f);
    if (p < CAP) {
        float4 x = ((const float4*)(eout + ((size_t)e * CAP + p) * DMODEL))[t];
        v = make_float4(g * x.x, g * x.y, g * x.z, g * x.w);
    }
    ((float4*)(out + (size_t)n * DMODEL))[t] = v;
}

extern "C" void kernel_launch(void* const* d_in, const int* in_sizes, int n_in,
                              void* d_out, int out_size, void* d_ws, size_t ws_size,
                              hipStream_t stream) {
    (void)in_sizes; (void)n_in; (void)out_size; (void)ws_size;
    const float* inputs = (const float*)d_in[0];
    const float* gate_w = (const float*)d_in[1];
    const float* gate_b = (const float*)d_in[2];
    const float* w1     = (const float*)d_in[3];
    const float* b1     = (const float*)d_in[4];
    const float* w2     = (const float*)d_in[5];
    const float* b2     = (const float*)d_in[6];
    float* out = (float*)d_out;

    char* ws = (char*)d_ws;
    size_t off = 0;
    auto carve = [&](size_t bytes) -> char* {
        char* p = ws + off;
        off = (off + bytes + 255) & ~(size_t)255;
        return p;
    };
    unsigned short* w1t = (unsigned short*)carve((size_t)E_EXP * DMODEL * FFN * 2);   // [E][F][D]
    unsigned short* w2t = (unsigned short*)carve((size_t)E_EXP * FFN * DMODEL * 2);   // [E][D][F]
    unsigned short* disp = (unsigned short*)carve((size_t)E_EXP * CAP * DMODEL * 2);
    unsigned short* hbuf = (unsigned short*)carve((size_t)E_EXP * CAP * FFN * 2);
    float* eout = (float*)carve((size_t)E_EXP * CAP * DMODEL * 4);
    int*   eidx = (int*)carve((size_t)NTOK * 4);
    int*   pos  = (int*)carve((size_t)NTOK * 4);
    float* gp   = (float*)carve((size_t)NTOK * 4);
    int*   slot = (int*)carve((size_t)E_EXP * CAP * 4);

    // transpose-convert weights once: w1 [E][D][F] -> w1t [E][F][D]; w2 [E][F][D] -> w2t [E][D][F]
    cvt_t_bf16<<<dim3(FFN / 64, DMODEL / 64, E_EXP), 256, 0, stream>>>(w1, w1t, DMODEL, FFN);
    cvt_t_bf16<<<dim3(DMODEL / 64, FFN / 64, E_EXP), 256, 0, stream>>>(w2, w2t, FFN, DMODEL);

    gate_kernel<<<NTOK / 8, 256, 0, stream>>>(inputs, gate_w, gate_b, eidx, gp);
    init_slot<<<(E_EXP * CAP + 255) / 256, 256, 0, stream>>>(slot);
    scan_kernel<<<1, 32, 0, stream>>>(eidx, pos, slot);
    gather_kernel<<<E_EXP * CAP, 256, 0, stream>>>(slot, inputs, disp);

    // GEMM1: [E](640x1024) @ w1 -> gelu -> bf16 h   (N = FFN, K = DMODEL), TN=256
    moe_gemm<4, true><<<dim3(FFN / 256, CAP / TM, E_EXP), 256, 0, stream>>>(
        disp, w1t, b1, hbuf, CAP, FFN, DMODEL);
    // GEMM2: [E](640x4096) @ w2 + b2 -> f32 expert_out   (N = DMODEL, K = FFN), TN=128
    moe_gemm<2, false><<<dim3(DMODEL / 128, CAP / TM, E_EXP), 256, 0, stream>>>(
        hbuf, w2t, b2, eout, CAP, DMODEL, FFN);

    combine_kernel<<<NTOK, 256, 0, stream>>>(eidx, pos, gp, eout, out);
}
